// HeteroGNN_67095979098386
// MI455X (gfx1250) — matrix operand order repaired and checked by reading, not compiled, not run on hardware
//
#include <hip/hip_runtime.h>
#include <stdint.h>

typedef __bf16 bf16_t;
typedef __attribute__((ext_vector_type(16))) __bf16 v16bf;
typedef __attribute__((ext_vector_type(8)))  float  v8f;

union Frag {
  v16bf v;
  uint32_t u[8];
};

__device__ __forceinline__ uint32_t pack2_bf16(float lo, float hi) {
  uint16_t a = __builtin_bit_cast(uint16_t, (__bf16)lo);
  uint16_t b = __builtin_bit_cast(uint16_t, (__bf16)hi);
  return ((uint32_t)b << 16) | (uint32_t)a;
}

// Fused 2-layer MLP:  D2 = (relu(A@W1 + b1))@W2 + b2
//  AMODE: 0 = gather A rows from h_bf via idx (ARITY args of 64 cols each)
//         1 = A = concat(h_bf row [64] , msg_in row [64])  (K1 == 128)
//         2 = A = direct bf16 matrix, row stride K1
//  EPI:   0 = scatter-add: split row of N2 into 64-wide chunks, atomicAdd into out_msg[idx]
//         1 = store rows [N2] to out_f32 and out_bf16
// One wave = one 16-row M tile; 4 waves (128 threads) per block.
// Weights are bf16, pre-transposed: w1t = [N1][K1], w2t = [N2][N1].
template<int K1, int N1, int N2, int AMODE, int EPI, int ARITY>
__global__ __launch_bounds__(128) void mlp2_kernel(
    const bf16_t* __restrict__ h_bf, const float* __restrict__ msg_in,
    const int* __restrict__ idx,
    const bf16_t* __restrict__ w1t, const float* __restrict__ b1,
    const bf16_t* __restrict__ w2t, const float* __restrict__ b2,
    float* __restrict__ out_msg, float* __restrict__ out_f32,
    bf16_t* __restrict__ out_bf16, int n_rows)
{
  __shared__ bf16_t lds_h1[4][16 * N1];   // per-wave private H1 tile (bf16)
  const int wave = threadIdx.x >> 5;
  const int lane = threadIdx.x & 31;
  const int lm = lane & 15;               // A row / B,D column within tile
  const int hi = lane >> 4;               // half-wave select
  const int row0 = (blockIdx.x * 4 + wave) * 16;
  if (row0 >= n_rows) return;             // wave-uniform: EXEC stays full for WMMA
  const int r = row0 + lm;

  // Per-lane gather bases: this lane's A row is fixed for the whole tile.
  const bf16_t* abase[(AMODE == 0) ? ARITY : 1];
  if constexpr (AMODE == 0) {
    #pragma unroll
    for (int j = 0; j < ARITY; ++j)
      abase[j] = h_bf + (size_t)idx[r * ARITY + j] * 64;
  }

  const v8f vzero = {0.f, 0.f, 0.f, 0.f, 0.f, 0.f, 0.f, 0.f};

  // ---------------- GEMM1: [16 x K1] @ [K1 x N1] ----------------
  v8f acc[N1 / 16];
  #pragma unroll
  for (int nt = 0; nt < N1 / 16; ++nt) acc[nt] = vzero;

  #pragma unroll
  for (int kb = 0; kb < K1 / 32; ++kb) {
    Frag a;
    #pragma unroll
    for (int v = 0; v < 8; ++v) {
      // A(16x32 bf16) layout: K = 16*(v>=4) + 8*hi + 2*(v&3) (+pair)
      const int k = kb * 32 + ((v >> 2) << 4) + hi * 8 + ((v & 3) << 1);
      if constexpr (AMODE == 0) {
        a.u[v] = *(const uint32_t*)(abase[k >> 6] + (k & 63));      // gathered
      } else if constexpr (AMODE == 1) {
        if (k < 64)
          a.u[v] = *(const uint32_t*)(h_bf + (size_t)r * 64 + k);
        else {
          const float* m = msg_in + (size_t)r * 64 + (k - 64);
          a.u[v] = pack2_bf16(m[0], m[1]);                          // f32 -> bf16 pair
        }
      } else {
        a.u[v] = *(const uint32_t*)(h_bf + (size_t)r * K1 + k);
      }
    }
    #pragma unroll
    for (int nt = 0; nt < N1 / 16; ++nt) {
      Frag b;
      const int n = nt * 16 + lm;
      #pragma unroll
      for (int v = 0; v < 8; ++v) {
        // B(32x16 bf16) layout: K = 16*(v>=4) + 8*hi + 2*(v&3) (+pair); W1^T row-major [N1][K1]
        const int k = kb * 32 + ((v >> 2) << 4) + hi * 8 + ((v & 3) << 1);
        b.u[v] = *(const uint32_t*)(w1t + (size_t)n * K1 + k);
      }
      acc[nt] = __builtin_amdgcn_wmma_f32_16x16x32_bf16(
          false, a.v, false, b.v, (short)0, acc[nt], false, false);
    }
  }

  // bias + ReLU -> per-wave LDS tile [16][N1] bf16 (relayout D -> A via LDS)
  bf16_t* my = &lds_h1[wave][0];
  #pragma unroll
  for (int nt = 0; nt < N1 / 16; ++nt) {
    const int n = nt * 16 + lm;
    const float bias = b1[n];
    #pragma unroll
    for (int e = 0; e < 8; ++e) {
      float val = acc[nt][e] + bias;               // D element (m = 8*hi+e, n)
      val = val > 0.f ? val : 0.f;
      my[(hi * 8 + e) * N1 + n] = (bf16_t)val;
    }
  }
  // same-wave LDS ops are in-order; each wave owns its region -> no barrier

  // ---------------- GEMM2: [16 x N1] @ [N1 x N2] ----------------
  v8f acc2[N2 / 16];
  #pragma unroll
  for (int nt = 0; nt < N2 / 16; ++nt) acc2[nt] = vzero;

  #pragma unroll
  for (int kb = 0; kb < N1 / 32; ++kb) {
    Frag a;
    #pragma unroll
    for (int v = 0; v < 8; ++v) {
      const int k = kb * 32 + ((v >> 2) << 4) + hi * 8 + ((v & 3) << 1);
      a.u[v] = *(const uint32_t*)(my + lm * N1 + k);               // ds_load_b32
    }
    #pragma unroll
    for (int nt = 0; nt < N2 / 16; ++nt) {
      Frag b;
      const int n = nt * 16 + lm;
      #pragma unroll
      for (int v = 0; v < 8; ++v) {
        const int k = kb * 32 + ((v >> 2) << 4) + hi * 8 + ((v & 3) << 1);
        b.u[v] = *(const uint32_t*)(w2t + (size_t)n * N1 + k);
      }
      acc2[nt] = __builtin_amdgcn_wmma_f32_16x16x32_bf16(
          false, a.v, false, b.v, (short)0, acc2[nt], false, false);
    }
  }

  // ---------------- epilogue ----------------
  #pragma unroll
  for (int nt = 0; nt < N2 / 16; ++nt) {
    const int n = nt * 16 + lm;
    const float bias = b2[n];
    if constexpr (EPI == 0) {
      const int j = n >> 6;            // which argument slot (uniform per nt)
      const int c = n & 63;
      #pragma unroll
      for (int e = 0; e < 8; ++e) {
        const int rr = row0 + hi * 8 + e;
        const int obj = idx[rr * ARITY + j];
        atomicAdd(out_msg + (size_t)obj * 64 + c, acc2[nt][e] + bias);
      }
    } else {
      #pragma unroll
      for (int e = 0; e < 8; ++e) {
        const int rr = row0 + hi * 8 + e;
        const float val = acc2[nt][e] + bias;
        out_f32[(size_t)rr * N2 + n] = val;
        out_bf16[(size_t)rr * N2 + n] = (bf16_t)val;
      }
    }
  }
}

// Convert f32 weight [Kin][N] -> bf16 transposed+padded [N][Kpad]
__global__ __launch_bounds__(256) void wt_kernel(const float* __restrict__ in,
    bf16_t* __restrict__ out, int Kin, int Kpad, int N) {
  int t = blockIdx.x * 256 + threadIdx.x;
  if (t >= N * Kpad) return;
  int n = t / Kpad, k = t - n * Kpad;
  out[t] = (k < Kin) ? (bf16_t)in[k * N + n] : (bf16_t)0.f;
}

// x_obj f32 [n,7] -> bf16 zero-padded [n,32]  (encoder K padded to one WMMA step)
__global__ __launch_bounds__(256) void xpad_kernel(const float* __restrict__ in,
    bf16_t* __restrict__ out, int n_rows) {
  int t = blockIdx.x * 256 + threadIdx.x;
  if (t >= n_rows * 32) return;
  int rr = t >> 5, k = t & 31;
  out[t] = (k < 7) ? (bf16_t)in[rr * 7 + k] : (bf16_t)0.f;
}

__global__ __launch_bounds__(256) void zero_kernel(float* __restrict__ p, int n) {
  int i = blockIdx.x * 256 + threadIdx.x;
  if (i < n) p[i] = 0.f;
}

// global_add_pool: LDS [64 graphs][64] accumulation (batch is sorted), then global atomics
__global__ __launch_bounds__(256) void pool_kernel(const float* __restrict__ h,
    const int* __restrict__ batch, float* __restrict__ out, int n_obj) {
  __shared__ float acc[64 * 64];
  for (int i = threadIdx.x; i < 64 * 64; i += 256) acc[i] = 0.f;
  __syncthreads();
  const long long start = (long long)blockIdx.x * 1024;
  long long end = start + 1024;
  if (end > n_obj) end = n_obj;
  for (long long f = start * 64 + threadIdx.x; f < end * 64; f += 256) {
    const int obj = (int)(f >> 6);
    const int c = (int)(f & 63);
    atomicAdd(&acc[batch[obj] * 64 + c], h[f]);   // ds_add_f32
  }
  __syncthreads();
  for (int i = threadIdx.x; i < 64 * 64; i += 256) {
    const float v = acc[i];
    if (v != 0.f) atomicAdd(out + i, v);
  }
}

static inline int cdiv(long long a, long long b) { return (int)((a + b - 1) / b); }

extern "C" void kernel_launch(void* const* d_in, const int* in_sizes, int n_in,
                              void* d_out, int out_size, void* d_ws, size_t ws_size,
                              hipStream_t stream) {
  (void)in_sizes; (void)n_in; (void)out_size; (void)ws_size;
  constexpr int N_OBJ = 100000, N_CLEAR = 200000, N_ON = 500000, N_AT = 200000;

  const float* x_obj     = (const float*)d_in[0];
  const int*   idx_clear = (const int*)d_in[4];
  const int*   idx_on    = (const int*)d_in[5];
  const int*   idx_at    = (const int*)d_in[6];
  const int*   batch     = (const int*)d_in[7];
  // params in setup_inputs() insertion order: enc_obj, obj_update, fanout_{clear,on,at}; each w1,b1,w2,b2
  const float* enc_w1 = (const float*)d_in[8];   const float* enc_b1 = (const float*)d_in[9];
  const float* enc_w2 = (const float*)d_in[10];  const float* enc_b2 = (const float*)d_in[11];
  const float* upd_w1 = (const float*)d_in[12];  const float* upd_b1 = (const float*)d_in[13];
  const float* upd_w2 = (const float*)d_in[14];  const float* upd_b2 = (const float*)d_in[15];
  const float* cl_w1  = (const float*)d_in[16];  const float* cl_b1  = (const float*)d_in[17];
  const float* cl_w2  = (const float*)d_in[18];  const float* cl_b2  = (const float*)d_in[19];
  const float* on_w1  = (const float*)d_in[20];  const float* on_b1  = (const float*)d_in[21];
  const float* on_w2  = (const float*)d_in[22];  const float* on_b2  = (const float*)d_in[23];
  const float* at_w1  = (const float*)d_in[24];  const float* at_b1  = (const float*)d_in[25];
  const float* at_w2  = (const float*)d_in[26];  const float* at_b2  = (const float*)d_in[27];

  char* ws = (char*)d_ws;
  size_t off = 0;
  auto alloc = [&](size_t bytes) -> char* {
    char* p = ws + off;
    off = (off + bytes + 255) & ~(size_t)255;
    return p;
  };
  bf16_t* h_bf   = (bf16_t*)alloc((size_t)N_OBJ * 64 * 2);
  float*  h_f32  = (float*) alloc((size_t)N_OBJ * 64 * 4);
  float*  msg    = (float*) alloc((size_t)N_OBJ * 64 * 4);
  bf16_t* xpad   = (bf16_t*)alloc((size_t)N_OBJ * 32 * 2);
  bf16_t* enc_w1t = (bf16_t*)alloc(64 * 32 * 2);
  bf16_t* enc_w2t = (bf16_t*)alloc(64 * 64 * 2);
  bf16_t* upd_w1t = (bf16_t*)alloc(128 * 128 * 2);
  bf16_t* upd_w2t = (bf16_t*)alloc(64 * 128 * 2);
  bf16_t* cl_w1t  = (bf16_t*)alloc(64 * 64 * 2);
  bf16_t* cl_w2t  = (bf16_t*)alloc(64 * 64 * 2);
  bf16_t* on_w1t  = (bf16_t*)alloc(128 * 128 * 2);
  bf16_t* on_w2t  = (bf16_t*)alloc(128 * 128 * 2);
  bf16_t* at_w1t  = (bf16_t*)alloc(192 * 192 * 2);
  bf16_t* at_w2t  = (bf16_t*)alloc(192 * 192 * 2);

  // ---- weight prep (bf16, transposed [N][Kpad]) ----
  wt_kernel<<<cdiv(64 * 32, 256),   256, 0, stream>>>(enc_w1, enc_w1t, 7,   32,  64);
  wt_kernel<<<cdiv(64 * 64, 256),   256, 0, stream>>>(enc_w2, enc_w2t, 64,  64,  64);
  wt_kernel<<<cdiv(128 * 128, 256), 256, 0, stream>>>(upd_w1, upd_w1t, 128, 128, 128);
  wt_kernel<<<cdiv(64 * 128, 256),  256, 0, stream>>>(upd_w2, upd_w2t, 128, 128, 64);
  wt_kernel<<<cdiv(64 * 64, 256),   256, 0, stream>>>(cl_w1,  cl_w1t,  64,  64,  64);
  wt_kernel<<<cdiv(64 * 64, 256),   256, 0, stream>>>(cl_w2,  cl_w2t,  64,  64,  64);
  wt_kernel<<<cdiv(128 * 128, 256), 256, 0, stream>>>(on_w1,  on_w1t,  128, 128, 128);
  wt_kernel<<<cdiv(128 * 128, 256), 256, 0, stream>>>(on_w2,  on_w2t,  128, 128, 128);
  wt_kernel<<<cdiv(192 * 192, 256), 256, 0, stream>>>(at_w1,  at_w1t,  192, 192, 192);
  wt_kernel<<<cdiv(192 * 192, 256), 256, 0, stream>>>(at_w2,  at_w2t,  192, 192, 192);
  xpad_kernel<<<cdiv((long long)N_OBJ * 32, 256), 256, 0, stream>>>(x_obj, xpad, N_OBJ);

  // ---- encoder: relu(x@W1+b1)@W2+b2, K padded 7->32 ----
  mlp2_kernel<32, 64, 64, 2, 1, 1><<<cdiv(N_OBJ, 64), 128, 0, stream>>>(
      xpad, nullptr, nullptr, enc_w1t, enc_b1, enc_w2t, enc_b2,
      nullptr, h_f32, h_bf, N_OBJ);

  // ---- 3 message-passing layers ----
  for (int layer = 0; layer < 3; ++layer) {
    zero_kernel<<<cdiv((long long)N_OBJ * 64, 256), 256, 0, stream>>>(msg, N_OBJ * 64);
    mlp2_kernel<64, 64, 64, 0, 0, 1><<<cdiv(N_CLEAR, 64), 128, 0, stream>>>(
        h_bf, nullptr, idx_clear, cl_w1t, cl_b1, cl_w2t, cl_b2,
        msg, nullptr, nullptr, N_CLEAR);
    mlp2_kernel<128, 128, 128, 0, 0, 2><<<cdiv(N_ON, 64), 128, 0, stream>>>(
        h_bf, nullptr, idx_on, on_w1t, on_b1, on_w2t, on_b2,
        msg, nullptr, nullptr, N_ON);
    mlp2_kernel<192, 192, 192, 0, 0, 3><<<cdiv(N_AT, 64), 128, 0, stream>>>(
        h_bf, nullptr, idx_at, at_w1t, at_b1, at_w2t, at_b2,
        msg, nullptr, nullptr, N_AT);
    // obj_update: concat(h, msg) -> relu(@W1+b1) -> @W2+b2; safe in-place (row-local)
    mlp2_kernel<128, 128, 64, 1, 1, 1><<<cdiv(N_OBJ, 64), 128, 0, stream>>>(
        h_bf, msg, nullptr, upd_w1t, upd_b1, upd_w2t, upd_b2,
        nullptr, h_f32, h_bf, N_OBJ);
  }

  // ---- global_add_pool ----
  zero_kernel<<<16, 256, 0, stream>>>((float*)d_out, 64 * 64);
  pool_kernel<<<cdiv(N_OBJ, 1024), 256, 0, stream>>>(h_f32, batch, (float*)d_out, N_OBJ);
}
